// GCN_69063074120331
// MI455X (gfx1250) — matrix operand order repaired and checked by reading, not compiled, not run on hardware
//
#include <hip/hip_runtime.h>
#include <hip/hip_bf16.h>

// ---------------------------------------------------------------------------
// GCN forward (2x GCNConv + linear head) for MI455X (gfx1250, wave32).
// GEMMs run on v_wmma_f32_16x16x32_bf16; edge aggregation is an L2-resident
// f32 atomic scatter (h and agg both fit in the 192MB L2).
// ---------------------------------------------------------------------------

#define N_NODES 50000
#define N_EDGES 500000
#define FEAT    256            // F_IN == HID == 256
#define M_TILES (N_NODES / 16) // 3125, exact

typedef __attribute__((ext_vector_type(16))) __bf16 v16bf;
typedef __attribute__((ext_vector_type(8)))  __bf16 v8bf;
typedef __attribute__((ext_vector_type(8)))  float  v8f;

static __device__ __forceinline__ __bf16 to_bf16(float f) {
    return static_cast<__bf16>(f);
}

// ---------------- elementwise f32 -> bf16 convert --------------------------
__global__ void k_convert_bf16(const float* __restrict__ in,
                               __bf16* __restrict__ out, int n) {
    int i = blockIdx.x * blockDim.x + threadIdx.x;
    if (i < n) out[i] = to_bf16(in[i]);
}

// ---------------- W [K][N] f32  ->  Wt [N][K] bf16 -------------------------
__global__ void k_transpose_w(const float* __restrict__ W,
                              __bf16* __restrict__ Wt) {
    int n = blockIdx.x;      // 0..255 output column
    int k = threadIdx.x;     // 0..255
    Wt[n * FEAT + k] = to_bf16(W[k * FEAT + n]);
}

// ---------------- degree / dinv --------------------------------------------
__global__ void k_fill_deg(float* __restrict__ deg, int n) {
    int i = blockIdx.x * blockDim.x + threadIdx.x;
    if (i < n) deg[i] = 1.0f;                       // self-loop weight
}

__global__ void k_deg_accum(const int* __restrict__ dst,
                            const float* __restrict__ ew,
                            float* __restrict__ deg, int e) {
    int i = blockIdx.x * blockDim.x + threadIdx.x;
    if (i < e) atomicAdd(&deg[dst[i]], ew[i]);
}

__global__ void k_dinv(float* __restrict__ deg, int n) {
    int i = blockIdx.x * blockDim.x + threadIdx.x;
    if (i < n) deg[i] = rsqrtf(deg[i]);             // deg >= 1 always
}

// ---------------- GEMM: H[M,256] = A_bf16[M,256] @ Wt_bf16[256n][256k] -----
// One wave computes a 16x256 output tile: 16 v8f accumulators, 8 K-steps of
// 32, 128 v_wmma_f32_16x16x32_bf16 per wave. Wave-uniform early exit keeps
// EXEC all-ones around WMMA.
__global__ void __launch_bounds__(256)
k_gemm_bf16(const __bf16* __restrict__ A, const __bf16* __restrict__ Wt,
            float* __restrict__ H, int mtiles) {
    const int lane = threadIdx.x & 31;
    const int wid  = threadIdx.x >> 5;
    const int mt   = blockIdx.x * 8 + wid;
    if (mt >= mtiles) return;                       // uniform per wave

    const int half = lane >> 4;
    const int l16  = lane & 15;
    const __bf16* arow = A + (size_t)(mt * 16 + l16) * FEAT;

    v8f c[16];
#pragma unroll
    for (int n = 0; n < 16; ++n) c[n] = (v8f)(0.0f);

#pragma unroll
    for (int kk = 0; kk < 8; ++kk) {
        const int kbase = kk * 32;
        // A fragment: lanes 0-15 hold K 0-15, lanes 16-31 hold K 16-31 of this
        // K-chunk (matches B).
        v16bf a = *(const v16bf*)(arow + kbase + half * 16);

#pragma unroll
        for (int n = 0; n < 16; ++n) {
            // B fragment: lanes 0-15 hold K kbase..kbase+15 of column n*16+l16,
            // lanes 16-31 hold K kbase+16..kbase+31 (Wt is [N][K] bf16).
            v16bf b = *(const v16bf*)(Wt + (size_t)(n * 16 + l16) * FEAT +
                                      kbase + half * 16);
            c[n] = __builtin_amdgcn_wmma_f32_16x16x32_bf16(
                false, a, false, b, (short)0, c[n], false, false);
        }
    }

    // D layout: VGPR j -> row M = half*8 + j, col N = l16 (per 16x16 subtile)
#pragma unroll
    for (int n = 0; n < 16; ++n) {
#pragma unroll
        for (int j = 0; j < 8; ++j) {
            H[(size_t)(mt * 16 + half * 8 + j) * FEAT + n * 16 + l16] = c[n][j];
        }
    }
}

// ---------------- aggregation ----------------------------------------------
// Self-loop term folded into init: agg[i] = dinv[i]^2 * h[i]
__global__ void k_agg_init(const float* __restrict__ h,
                           const float* __restrict__ dinv,
                           float* __restrict__ agg, int nf) {
    int i = blockIdx.x * blockDim.x + threadIdx.x;
    if (i < nf) {
        float di = dinv[i >> 8];
        agg[i] = h[i] * di * di;
    }
}

// One wave per edge, features striped across lanes (coalesced, L2-resident).
__global__ void __launch_bounds__(256)
k_agg_edges(const int* __restrict__ src, const int* __restrict__ dst,
            const float* __restrict__ ew, const float* __restrict__ dinv,
            const float* __restrict__ h, float* __restrict__ agg, int e) {
    const int lane = threadIdx.x & 31;
    const int edge = (blockIdx.x * blockDim.x + threadIdx.x) >> 5;
    if (edge >= e) return;                          // uniform per wave
    const int s = src[edge];
    const int d = dst[edge];
    const float coeff = dinv[s] * ew[edge] * dinv[d];
    const float* hs = h + (size_t)s * FEAT;
    float* ad = agg + (size_t)d * FEAT;
#pragma unroll
    for (int f = 0; f < FEAT; f += 32)
        atomicAdd(&ad[f + lane], hs[f + lane] * coeff);
}

// bias + relu, emit bf16 for next GEMM
__global__ void k_bias_relu_bf16(const float* __restrict__ agg,
                                 const float* __restrict__ bias,
                                 __bf16* __restrict__ out, int nf) {
    int i = blockIdx.x * blockDim.x + threadIdx.x;
    if (i < nf) {
        float v = agg[i] + bias[i & (FEAT - 1)];
        out[i] = to_bf16(fmaxf(v, 0.0f));
    }
}

// ---------------- head: out[i] = relu(agg[i]+b2) . Wl + bl -----------------
__global__ void __launch_bounds__(256)
k_final(const float* __restrict__ agg, const float* __restrict__ b2,
        const float* __restrict__ Wl, const float* __restrict__ bl,
        float* __restrict__ out, int n) {
    const int lane = threadIdx.x & 31;
    const int node = (blockIdx.x * blockDim.x + threadIdx.x) >> 5;
    if (node >= n) return;
    const float* row = agg + (size_t)node * FEAT;
    float acc = 0.0f;
#pragma unroll
    for (int f = 0; f < FEAT; f += 32) {
        float v = fmaxf(row[f + lane] + b2[f + lane], 0.0f);
        acc += v * Wl[f + lane];
    }
#pragma unroll
    for (int m = 16; m >= 1; m >>= 1) acc += __shfl_xor(acc, m, 32);
    if (lane == 0) out[node] = acc + bl[0];
}

// ---------------------------------------------------------------------------
extern "C" void kernel_launch(void* const* d_in, const int* in_sizes, int n_in,
                              void* d_out, int out_size, void* d_ws, size_t ws_size,
                              hipStream_t stream) {
    const float* x   = (const float*)d_in[0];
    const int*   ei  = (const int*)d_in[1];
    const float* ew  = (const float*)d_in[2];
    const float* W1  = (const float*)d_in[3];
    const float* b1  = (const float*)d_in[4];
    const float* W2  = (const float*)d_in[5];
    const float* b2  = (const float*)d_in[6];
    const float* Wl  = (const float*)d_in[7];
    const float* bl  = (const float*)d_in[8];
    float* out = (float*)d_out;

    const int*  src = ei;
    const int*  dst = ei + N_EDGES;
    const int   NF  = N_NODES * FEAT;

    // workspace carve-out (~128.5 MB)
    char*  ws  = (char*)d_ws;
    size_t off = 0;
    auto carve = [&](size_t bytes) {
        void* p = ws + off;
        off = (off + bytes + 255) & ~(size_t)255;
        return p;
    };
    float*  dinv = (float*)carve((size_t)N_NODES * 4);
    float*  h    = (float*)carve((size_t)NF * 4);
    float*  agg  = (float*)carve((size_t)NF * 4);
    __bf16* xb   = (__bf16*)carve((size_t)NF * 2);   // aliased: x_bf16 / h_bf16
    __bf16* W1t  = (__bf16*)carve((size_t)FEAT * FEAT * 2);
    __bf16* W2t  = (__bf16*)carve((size_t)FEAT * FEAT * 2);

    const int TB = 256;
    const int gNF   = (NF + TB - 1) / TB;            // 50000
    const int gN    = (N_NODES + TB - 1) / TB;       // 196
    const int gE    = (N_EDGES + TB - 1) / TB;       // 1954
    const int gGemm = (M_TILES + 7) / 8;             // 391 blocks of 8 waves
    const int gEdgeW = (N_EDGES * 32 + TB - 1) / TB; // wave per edge
    const int gNodeW = (N_NODES * 32 + TB - 1) / TB; // wave per node

    // ---- normalization precompute ----
    k_fill_deg <<<gN, TB, 0, stream>>>(dinv, N_NODES);
    k_deg_accum<<<gE, TB, 0, stream>>>(dst, ew, dinv, N_EDGES);
    k_dinv     <<<gN, TB, 0, stream>>>(dinv, N_NODES);

    // ---- weight prep ----
    k_transpose_w<<<FEAT, FEAT, 0, stream>>>(W1, W1t);
    k_transpose_w<<<FEAT, FEAT, 0, stream>>>(W2, W2t);

    // ---- layer 1 ----
    k_convert_bf16<<<gNF, TB, 0, stream>>>(x, xb, NF);
    k_gemm_bf16   <<<gGemm, TB, 0, stream>>>(xb, W1t, h, M_TILES);
    k_agg_init    <<<gNF, TB, 0, stream>>>(h, dinv, agg, NF);
    k_agg_edges   <<<gEdgeW, TB, 0, stream>>>(src, dst, ew, dinv, h, agg, N_EDGES);
    k_bias_relu_bf16<<<gNF, TB, 0, stream>>>(agg, b1, xb, NF);   // xb now = h1_bf16

    // ---- layer 2 ----
    k_gemm_bf16   <<<gGemm, TB, 0, stream>>>(xb, W2t, h, M_TILES);
    k_agg_init    <<<gNF, TB, 0, stream>>>(h, dinv, agg, NF);
    k_agg_edges   <<<gEdgeW, TB, 0, stream>>>(src, dst, ew, dinv, h, agg, N_EDGES);

    // ---- head ----
    k_final<<<gNodeW, TB, 0, stream>>>(agg, b2, Wl, bl, out, N_NODES);
}
